// HierarchicalDiffAttentionEncoderLayer_51711406244190
// MI455X (gfx1250) — compile-verified, matrix operand order checked by reading, not run on hardware
//
#include <hip/hip_runtime.h>
#include <math.h>

// ---------------- problem constants ----------------
#define BB 4
#define SS 1024
#define DM 512
#define NH 8
#define DH 64
#define D2 1024   // 2*DM
#define FF 2048
#define MM 4096   // BB*SS

typedef __attribute__((ext_vector_type(16))) __bf16 v16bf;
typedef __attribute__((ext_vector_type(8)))  float  v8f;
typedef __attribute__((ext_vector_type(4)))  unsigned int u32x4;

union BF16Frag {
    v16bf v;
    u32x4 u[2];
    unsigned short s[16];
};

__device__ __forceinline__ unsigned short f2bf(float f) {
    union { float f; unsigned u; } c; c.f = f;
    return (unsigned short)((c.u + 0x7FFFu + ((c.u >> 16) & 1u)) >> 16);
}

// Load a 16x32 bf16 A-style fragment (also used for B of A@B^T GEMMs):
// lane m = l&15 owns one row; K offset = (l>=16 ? 8 : 0); elems 0..7 = K+0..7,
// elems 8..15 = K+16..23 (per CDNA5 16-bit A-matrix layout).
__device__ __forceinline__ v16bf load_frag(const unsigned short* __restrict__ base,
                                           int rowStride, int row, int colBase, int lane) {
    int koff = ((lane >> 4) << 3);
    const unsigned short* p = base + (size_t)row * rowStride + colBase + koff;
    BF16Frag f;
    f.u[0] = *reinterpret_cast<const u32x4*>(p);
    f.u[1] = *reinterpret_cast<const u32x4*>(p + 16);
    return f.v;
}

#define WMMA_BF16(Afrag, Bfrag, Cacc) \
    __builtin_amdgcn_wmma_f32_16x16x32_bf16(false, (Afrag), false, (Bfrag), (short)0, (Cacc), false, false)

// CDNA5 async global->LDS DMA (tracked by ASYNCcnt). 16 bytes per lane.
__device__ __forceinline__ void async_b128_to_lds(const unsigned short* g,
                                                  const unsigned short* lds) {
    unsigned ldsoff = (unsigned)(size_t)lds;                 // low 32 bits = LDS byte offset
    unsigned long long ga = (unsigned long long)(size_t)g;
    asm volatile("global_load_async_to_lds_b128 %0, %1, off"
                 :: "v"(ldsoff), "v"(ga) : "memory");
}

__device__ __forceinline__ void wait_asynccnt0() {
    asm volatile("s_wait_asynccnt 0x0" ::: "memory");
}

// Copy one 64x32-bf16 (64 rows x 64 bytes) tile: 256 16-byte chunks, 128 threads.
__device__ __forceinline__ void async_copy_tile64x32(const unsigned short* __restrict__ gbase,
                                                     int rowStride, int rowBase, int kk,
                                                     unsigned short* lds, int tid) {
#pragma unroll
    for (int i = 0; i < 2; ++i) {
        int c = tid + i * 128;
        int row = c >> 2, col = (c & 3) * 8;
        const unsigned short* g = gbase + (size_t)(rowBase + row) * rowStride + kk + col;
        async_b128_to_lds(g, lds + row * 32 + col);
    }
}

// ---------------- weight transpose + bf16 cast: wT[n*K+k] = bf16(w[k*N+n]) ----
__global__ void transpose_cast_kernel(const float* __restrict__ w,
                                      unsigned short* __restrict__ wT, int K, int N) {
    int idx = blockIdx.x * blockDim.x + threadIdx.x;
    if (idx >= K * N) return;
    int k = idx / N, n = idx % N;
    wT[(size_t)n * K + k] = f2bf(w[idx]);
}

// ---------------- RMSNorm -> bf16 ----------------
__global__ void rmsnorm_bf16_kernel(const float* __restrict__ x, const float* __restrict__ w,
                                    unsigned short* __restrict__ out) {
    __shared__ float red[128];
    int row = blockIdx.x, tid = threadIdx.x;
    const float* xr = x + (size_t)row * DM;
    float s = 0.f;
    for (int c = tid; c < DM; c += 128) { float v = xr[c]; s += v * v; }
    red[tid] = s; __syncthreads();
    for (int o = 64; o > 0; o >>= 1) {
        if (tid < o) red[tid] += red[tid + o];
        __syncthreads();
    }
    float rs = rsqrtf(red[0] / (float)DM + 1e-5f);
    for (int c = tid; c < DM; c += 128)
        out[(size_t)row * DM + c] = f2bf(xr[c] * rs * w[c]);
}

// ------- cooperative WMMA GEMM: 4 waves, 64x64 block tile, async double buffer.
// out = A[M,K](bf16) @ WT[N,K](bf16)^T (+bias[N]) (+addsrc[M,N]); f32 and/or bf16 out.
__global__ void __launch_bounds__(128) wmma_gemm_lds_kernel(
        const unsigned short* __restrict__ A, const unsigned short* __restrict__ WT,
        const float* __restrict__ bias, const float* __restrict__ addsrc,
        float* __restrict__ outF, unsigned short* __restrict__ outB,
        int Mn, int Nn, int Kn) {
    __shared__ alignas(16) unsigned short tiles[2][2][64 * 32];  // [buf][A/B][64 rows x 32 bf16]
    int tid = threadIdx.x;
    int lane = tid & 31, w = tid >> 5;
    int l15 = lane & 15, hw = lane >> 4;
    int blkRow = blockIdx.y * 64;
    int blkCol = blockIdx.x * 64;

    v8f acc[4];
#pragma unroll
    for (int t = 0; t < 4; ++t) {
        float bv = bias ? bias[blkCol + t * 16 + l15] : 0.f;
#pragma unroll
        for (int r = 0; r < 8; ++r) acc[t][r] = bv;
    }

    int nsteps = Kn / 32;
    // prologue: stage K-step 0 into buffer 0
    async_copy_tile64x32(A, Kn, blkRow, 0, &tiles[0][0][0], tid);
    async_copy_tile64x32(WT, Kn, blkCol, 0, &tiles[0][1][0], tid);

    for (int s = 0; s < nsteps; ++s) {
        wait_asynccnt0();      // my DMA chunks landed
        __syncthreads();       // everyone's chunks visible
        if (s + 1 < nsteps) {  // overlap next K-step DMA with this step's WMMAs
            int nb = (s + 1) & 1;
            async_copy_tile64x32(A, Kn, blkRow, (s + 1) * 32, &tiles[nb][0][0], tid);
            async_copy_tile64x32(WT, Kn, blkCol, (s + 1) * 32, &tiles[nb][1][0], tid);
        }
        const unsigned short* tA = &tiles[s & 1][0][0];
        const unsigned short* tB = &tiles[s & 1][1][0];
        v16bf a = load_frag(tA, 32, w * 16 + l15, 0, lane);
#pragma unroll
        for (int t = 0; t < 4; ++t) {
            v16bf b = load_frag(tB, 32, t * 16 + l15, 0, lane);
            acc[t] = WMMA_BF16(a, b, acc[t]);
        }
        // buffer s&1 is next overwritten by the DMA issued at step s+1 (K-step s+2), one barrier away:
        // all ds reads above complete (consumed by WMMA) before the next barrier.
    }

    int rowBase = blkRow + w * 16;
#pragma unroll
    for (int t = 0; t < 4; ++t) {
        int col = blkCol + t * 16 + l15;
#pragma unroll
        for (int r = 0; r < 8; ++r) {
            int row = rowBase + r + 8 * hw;
            float v = acc[t][r];
            if (addsrc) v += addsrc[(size_t)row * Nn + col];
            if (outF) outF[(size_t)row * Nn + col] = v;
            if (outB) outB[(size_t)row * Nn + col] = f2bf(v);
        }
    }
}

// ---------------- neighbor affinity: loga[b,i] -------------------------------
__global__ void __launch_bounds__(32) neighbor_kernel(const float* __restrict__ Q,
                                                      const float* __restrict__ Kn,
                                                      float* __restrict__ loga) {
    int lane = threadIdx.x;
    int idx = blockIdx.x;
    int b = idx / (SS - 1), i = idx % (SS - 1);
    const float* q0 = Q + (size_t)(b * SS + i) * DM;
    const float* q1 = q0 + DM;
    const float* k0 = Kn + (size_t)(b * SS + i) * DM;
    const float* k1 = k0 + DM;
    float fwd = 0.f, bwd = 0.f;
    for (int d = lane; d < DM; d += 32) { fwd += q0[d] * k1[d]; bwd += q1[d] * k0[d]; }
#pragma unroll
    for (int o = 16; o > 0; o >>= 1) {
        fwd += __shfl_xor(fwd, o);
        bwd += __shfl_xor(bwd, o);
    }
    if (lane == 0) {
        float sf = fwd / 512.0f, sb = bwd / 512.0f;
        float a = 0.5f * (1.f / (1.f + __expf(-sf)) + 1.f / (1.f + __expf(-sb)));
        loga[b * SS + i] = __logf(a);
    }
}

// ---------------- prefix scan: L[b,i] = sum_{j<i} loga[b,j] ------------------
__global__ void scan_kernel(const float* __restrict__ loga, float* __restrict__ L) {
    __shared__ float sb[SS];
    int b = blockIdx.x, tid = threadIdx.x;
    float v = (tid == 0) ? 0.f : loga[b * SS + tid - 1];
    for (int o = 1; o < SS; o <<= 1) {
        sb[tid] = v; __syncthreads();
        if (tid >= o) v += sb[tid - o];
        __syncthreads();
    }
    L[b * SS + tid] = v;
}

// ---------------- attention pass 1: per-row softmax stats (both branches) ----
__global__ void __launch_bounds__(32) attn_stats_kernel(
        const unsigned short* __restrict__ qp, const unsigned short* __restrict__ kp,
        const int* __restrict__ pad,
        float* __restrict__ m1s, float* __restrict__ z1s,
        float* __restrict__ m2s, float* __restrict__ z2s) {
    int lane = threadIdx.x;
    int qt = blockIdx.x, h = blockIdx.y, b = blockIdx.z;
    int qbase = qt * 16;
    int l15 = lane & 15, hw = lane >> 4;
    int c1 = h * DH, c2 = DM + h * DH;
    int arow = b * SS + qbase + l15;
    v16bf a10 = load_frag(qp, D2, arow, c1, lane);
    v16bf a11 = load_frag(qp, D2, arow, c1 + 32, lane);
    v16bf a20 = load_frag(qp, D2, arow, c2, lane);
    v16bf a21 = load_frag(qp, D2, arow, c2 + 32, lane);
    float m1[8], z1[8], m2[8], z2[8];
#pragma unroll
    for (int r = 0; r < 8; ++r) { m1[r] = -3.0e38f; z1[r] = 0.f; m2[r] = -3.0e38f; z2[r] = 0.f; }
    const float scale = 0.125f;  // 1/sqrt(64)
    for (int kt = 0; kt < SS / 16; ++kt) {
        int krow = b * SS + kt * 16 + l15;
        v16bf b10 = load_frag(kp, D2, krow, c1, lane);
        v16bf b11 = load_frag(kp, D2, krow, c1 + 32, lane);
        v16bf b20 = load_frag(kp, D2, krow, c2, lane);
        v16bf b21 = load_frag(kp, D2, krow, c2 + 32, lane);
        v8f s1 = {0.f, 0.f, 0.f, 0.f, 0.f, 0.f, 0.f, 0.f};
        v8f s2 = s1;
        s1 = WMMA_BF16(a10, b10, s1);
        s1 = WMMA_BF16(a11, b11, s1);
        s2 = WMMA_BF16(a20, b20, s2);
        s2 = WMMA_BF16(a21, b21, s2);
        int kcol = kt * 16 + l15;
        bool mo = (pad[b * SS + kcol] == 0);
#pragma unroll
        for (int r = 0; r < 8; ++r) {
            float v1 = mo ? -1e9f : s1[r] * scale;
            float v2 = mo ? -1e9f : s2[r] * scale;
            float t1 = v1;
            t1 = fmaxf(t1, __shfl_xor(t1, 1));
            t1 = fmaxf(t1, __shfl_xor(t1, 2));
            t1 = fmaxf(t1, __shfl_xor(t1, 4));
            t1 = fmaxf(t1, __shfl_xor(t1, 8));
            float mn1 = fmaxf(m1[r], t1);
            float e1 = __expf(v1 - mn1);
            e1 += __shfl_xor(e1, 1); e1 += __shfl_xor(e1, 2);
            e1 += __shfl_xor(e1, 4); e1 += __shfl_xor(e1, 8);
            z1[r] = z1[r] * __expf(m1[r] - mn1) + e1;
            m1[r] = mn1;
            float t2 = v2;
            t2 = fmaxf(t2, __shfl_xor(t2, 1));
            t2 = fmaxf(t2, __shfl_xor(t2, 2));
            t2 = fmaxf(t2, __shfl_xor(t2, 4));
            t2 = fmaxf(t2, __shfl_xor(t2, 8));
            float mn2 = fmaxf(m2[r], t2);
            float e2 = __expf(v2 - mn2);
            e2 += __shfl_xor(e2, 1); e2 += __shfl_xor(e2, 2);
            e2 += __shfl_xor(e2, 4); e2 += __shfl_xor(e2, 8);
            z2[r] = z2[r] * __expf(m2[r] - mn2) + e2;
            m2[r] = mn2;
        }
    }
    if (l15 == 0) {
#pragma unroll
        for (int r = 0; r < 8; ++r) {
            int qrow = qbase + r + 8 * hw;
            size_t so = (size_t)(b * NH + h) * SS + qrow;
            m1s[so] = m1[r]; z1s[so] = z1[r];
            m2s[so] = m2[r]; z2s[so] = z2[r];
        }
    }
}

// ---------------- attention pass 2: P = C*(p1 - lam*p2), ctx = P @ V (WMMA) --
__global__ void __launch_bounds__(32) attn_ctx_kernel(
        const unsigned short* __restrict__ qp, const unsigned short* __restrict__ kp,
        const unsigned short* __restrict__ vbf, const int* __restrict__ pad,
        const float* __restrict__ m1s, const float* __restrict__ z1s,
        const float* __restrict__ m2s, const float* __restrict__ z2s,
        const float* __restrict__ Lb, const float* __restrict__ lamp,
        float* __restrict__ ctx) {
    __shared__ alignas(16) unsigned short Pbuf[16 * 32];  // 16x32 bf16 P tile, row-major
    int lane = threadIdx.x;
    int qt = blockIdx.x, h = blockIdx.y, b = blockIdx.z;
    int qbase = qt * 16;
    int l15 = lane & 15, hw = lane >> 4;
    int c1 = h * DH, c2 = DM + h * DH;
    float lam = lamp[0];
    int arow = b * SS + qbase + l15;
    v16bf a10 = load_frag(qp, D2, arow, c1, lane);
    v16bf a11 = load_frag(qp, D2, arow, c1 + 32, lane);
    v16bf a20 = load_frag(qp, D2, arow, c2, lane);
    v16bf a21 = load_frag(qp, D2, arow, c2 + 32, lane);
    float m1r[8], z1r[8], m2r[8], z2r[8], Lq[8];
#pragma unroll
    for (int r = 0; r < 8; ++r) {
        int qrow = qbase + r + 8 * hw;
        size_t so = (size_t)(b * NH + h) * SS + qrow;
        m1r[r] = m1s[so]; z1r[r] = 1.0f / z1s[so];
        m2r[r] = m2s[so]; z2r[r] = 1.0f / z2s[so];
        Lq[r] = Lb[b * SS + qrow];
    }
    v8f acc[4];
#pragma unroll
    for (int t = 0; t < 4; ++t)
#pragma unroll
        for (int r = 0; r < 8; ++r) acc[t][r] = 0.f;
    const float scale = 0.125f;
    for (int k32 = 0; k32 < SS / 32; ++k32) {
#pragma unroll
        for (int sub = 0; sub < 2; ++sub) {
            int kbase = k32 * 32 + sub * 16;
            int krow = b * SS + kbase + l15;
            v16bf b10 = load_frag(kp, D2, krow, c1, lane);
            v16bf b11 = load_frag(kp, D2, krow, c1 + 32, lane);
            v16bf b20 = load_frag(kp, D2, krow, c2, lane);
            v16bf b21 = load_frag(kp, D2, krow, c2 + 32, lane);
            v8f s1 = {0.f, 0.f, 0.f, 0.f, 0.f, 0.f, 0.f, 0.f};
            v8f s2 = s1;
            s1 = WMMA_BF16(a10, b10, s1);
            s1 = WMMA_BF16(a11, b11, s1);
            s2 = WMMA_BF16(a20, b20, s2);
            s2 = WMMA_BF16(a21, b21, s2);
            int kcol = kbase + l15;
            bool mo = (pad[b * SS + kcol] == 0);
            float Lk = Lb[b * SS + kcol];
#pragma unroll
            for (int r = 0; r < 8; ++r) {
                float p = 0.f;
                if (!mo) {
                    float e1 = __expf(s1[r] * scale - m1r[r]) * z1r[r];
                    float e2 = __expf(s2[r] * scale - m2r[r]) * z2r[r];
                    p = __expf(-fabsf(Lq[r] - Lk)) * (e1 - lam * e2);
                }
                Pbuf[(r + 8 * hw) * 32 + sub * 16 + l15] = f2bf(p);
            }
        }
        __syncthreads();
        // reload P as a 16x32 A-fragment from LDS (layout conversion)
        int koff = hw * 8;
        BF16Frag pf;
        pf.u[0] = *reinterpret_cast<const u32x4*>(&Pbuf[l15 * 32 + koff]);
        pf.u[1] = *reinterpret_cast<const u32x4*>(&Pbuf[l15 * 32 + koff + 16]);
        __syncthreads();
        // V B-fragments: B[k][n] = V[k32*32+k, h*64 + 16t + n] (strided gather)
#pragma unroll
        for (int t = 0; t < 4; ++t) {
            BF16Frag vf;
            int col = h * DH + t * 16 + l15;
#pragma unroll
            for (int j = 0; j < 8; ++j) {
                vf.s[j]     = vbf[(size_t)(b * SS + k32 * 32 + koff + j) * DM + col];
                vf.s[j + 8] = vbf[(size_t)(b * SS + k32 * 32 + koff + 16 + j) * DM + col];
            }
            acc[t] = WMMA_BF16(pf.v, vf.v, acc[t]);
        }
    }
#pragma unroll
    for (int t = 0; t < 4; ++t) {
        int col = h * DH + t * 16 + l15;
#pragma unroll
        for (int r = 0; r < 8; ++r) {
            int qrow = qbase + r + 8 * hw;
            ctx[(size_t)(b * SS + qrow) * DM + col] = acc[t][r];
        }
    }
}

// ---------------- GroupNorm over (d_head*S) per (b,head) -> bf16 * 0.8 ------
__global__ void groupnorm_kernel(const float* __restrict__ ctx,
                                 const float* __restrict__ gamma, const float* __restrict__ beta,
                                 unsigned short* __restrict__ out) {
    __shared__ float r1[256], r2[256];
    int bh = blockIdx.x;
    int b = bh / NH, h = bh % NH;
    int tid = threadIdx.x;
    float s = 0.f, sq = 0.f;
    for (int i = tid; i < DH * SS; i += 256) {
        int sidx = i >> 6, d = i & 63;
        float v = ctx[(size_t)(b * SS + sidx) * DM + h * DH + d];
        s += v; sq += v * v;
    }
    r1[tid] = s; r2[tid] = sq; __syncthreads();
    for (int o = 128; o > 0; o >>= 1) {
        if (tid < o) { r1[tid] += r1[tid + o]; r2[tid] += r2[tid + o]; }
        __syncthreads();
    }
    float inv = 1.0f / (float)(DH * SS);
    float mean = r1[0] * inv;
    float var = r2[0] * inv - mean * mean;
    float rstd = rsqrtf(var + 1e-5f);
    for (int i = tid; i < DH * SS; i += 256) {
        int sidx = i >> 6, d = i & 63;
        int c = h * DH + d;
        size_t o2 = (size_t)(b * SS + sidx) * DM + c;
        float v = ctx[o2];
        out[o2] = f2bf(((v - mean) * rstd * gamma[c] + beta[c]) * 0.8f);  // * (1 - lambda_init)
    }
}

// ---------------- fused SwiGLU dual-GEMM: act = silu(A@Wg) * (A@W1) -> bf16 --
__global__ void __launch_bounds__(32) swiglu_gemm_kernel(
        const unsigned short* __restrict__ A, const unsigned short* __restrict__ WgT,
        const unsigned short* __restrict__ W1T, unsigned short* __restrict__ out,
        int Mn, int Nn, int Kn) {
    int lane = threadIdx.x;
    int l15 = lane & 15, hw = lane >> 4;
    int rowBase = blockIdx.y * 16;
    int colBase = blockIdx.x * 64;
    v8f g[4], u[4];
#pragma unroll
    for (int t = 0; t < 4; ++t)
#pragma unroll
        for (int r = 0; r < 8; ++r) { g[t][r] = 0.f; u[t][r] = 0.f; }
    int arow = rowBase + l15;
    for (int kk = 0; kk < Kn; kk += 32) {
        v16bf a = load_frag(A, Kn, arow, kk, lane);
#pragma unroll
        for (int t = 0; t < 4; ++t) {
            v16bf bg = load_frag(WgT, Kn, colBase + t * 16 + l15, kk, lane);
            v16bf bu = load_frag(W1T, Kn, colBase + t * 16 + l15, kk, lane);
            g[t] = WMMA_BF16(a, bg, g[t]);
            u[t] = WMMA_BF16(a, bu, u[t]);
        }
    }
#pragma unroll
    for (int t = 0; t < 4; ++t) {
        int col = colBase + t * 16 + l15;
#pragma unroll
        for (int r = 0; r < 8; ++r) {
            int row = rowBase + r + 8 * hw;
            float gg = g[t][r];
            float act = (gg / (1.f + __expf(-gg))) * u[t][r];
            out[(size_t)row * Nn + col] = f2bf(act);
        }
    }
}

// ========================= host-side orchestration ==========================
extern "C" void kernel_launch(void* const* d_in, const int* in_sizes, int n_in,
                              void* d_out, int out_size, void* d_ws, size_t ws_size,
                              hipStream_t stream) {
    (void)in_sizes; (void)n_in; (void)out_size; (void)ws_size;
    const float* x     = (const float*)d_in[0];
    const int*   pad   = (const int*)d_in[1];
    const float* lamp  = (const float*)d_in[2];
    const float* wq_w  = (const float*)d_in[3];
    const float* wq_b  = (const float*)d_in[4];
    const float* wk_w  = (const float*)d_in[5];
    const float* wk_b  = (const float*)d_in[6];
    const float* wv_w  = (const float*)d_in[7];
    const float* wv_b  = (const float*)d_in[8];
    const float* out_w = (const float*)d_in[9];
    const float* out_b = (const float*)d_in[10];
    const float* gng   = (const float*)d_in[11];
    const float* gnb   = (const float*)d_in[12];
    const float* aq_w  = (const float*)d_in[13];
    const float* aq_b  = (const float*)d_in[14];
    const float* ak_w  = (const float*)d_in[15];
    const float* ak_b  = (const float*)d_in[16];
    const float* wg_w  = (const float*)d_in[17];
    const float* w1_w  = (const float*)d_in[18];
    const float* w2_w  = (const float*)d_in[19];
    const float* n1_w  = (const float*)d_in[20];
    const float* n2_w  = (const float*)d_in[21];
    float* out = (float*)d_out;

    char* ws = (char*)d_ws;
    const size_t MBy = 1024u * 1024u;
    unsigned short* xn_bf  = (unsigned short*)(ws + 0);          // 4 MB
    unsigned short* qp_bf  = (unsigned short*)(ws + 4 * MBy);    // 8 MB
    unsigned short* kp_bf  = (unsigned short*)(ws + 12 * MBy);   // 8 MB
    unsigned short* v_bf   = (unsigned short*)(ws + 20 * MBy);   // 4 MB
    float* aQ   = (float*)(ws + 24 * MBy);                       // 8 MB
    float* aK   = (float*)(ws + 32 * MBy);                       // 8 MB
    float* loga = (float*)(ws + 40 * MBy);                       // 16 KB
    float* Lb   = (float*)(ws + 40 * MBy + 64 * 1024);           // 16 KB
    float* m1s  = (float*)(ws + 40 * MBy + 128 * 1024);          // 128 KB each
    float* z1s  = (float*)(ws + 40 * MBy + 256 * 1024);
    float* m2s  = (float*)(ws + 40 * MBy + 384 * 1024);
    float* z2s  = (float*)(ws + 40 * MBy + 512 * 1024);
    float* ctx  = (float*)(ws + 41 * MBy);                       // 8 MB
    unsigned short* gn_bf  = (unsigned short*)(ws + 49 * MBy);   // 4 MB
    float* hbuf = (float*)(ws + 53 * MBy);                       // 8 MB
    unsigned short* hn_bf  = (unsigned short*)(ws + 61 * MBy);   // 4 MB
    unsigned short* act_bf = (unsigned short*)(ws + 65 * MBy);   // 16 MB
    unsigned short* wqT  = (unsigned short*)(ws + 81 * MBy);     // [1024][512]
    unsigned short* wkT  = (unsigned short*)(ws + 82 * MBy);     // [1024][512]
    unsigned short* wvT  = (unsigned short*)(ws + 83 * MBy);     // [512][512]
    unsigned short* outT = (unsigned short*)(ws + 83 * MBy + 512 * 1024);
    unsigned short* aqT  = (unsigned short*)(ws + 84 * MBy);
    unsigned short* akT  = (unsigned short*)(ws + 84 * MBy + 512 * 1024);
    unsigned short* wgT  = (unsigned short*)(ws + 85 * MBy);     // [2048][512]
    unsigned short* w1T  = (unsigned short*)(ws + 87 * MBy);     // [2048][512]
    unsigned short* w2T  = (unsigned short*)(ws + 89 * MBy);     // [512][2048]

    // ---- weight transpose/cast to bf16 ----
    auto tc = [&](const float* w, unsigned short* wT, int K, int N) {
        int n = K * N;
        transpose_cast_kernel<<<(n + 255) / 256, 256, 0, stream>>>(w, wT, K, N);
    };
    tc(wq_w, wqT, DM, D2);
    tc(wk_w, wkT, DM, D2);
    tc(wv_w, wvT, DM, DM);
    tc(out_w, outT, DM, DM);
    tc(aq_w, aqT, DM, DM);
    tc(ak_w, akT, DM, DM);
    tc(wg_w, wgT, DM, FF);
    tc(w1_w, w1T, DM, FF);
    tc(w2_w, w2T, FF, DM);

    auto gemm = [&](const unsigned short* A, const unsigned short* WT, const float* bias,
                    const float* addsrc, float* oF, unsigned short* oB, int Mn, int Nn, int Kn) {
        dim3 g(Nn / 64, Mn / 64);
        wmma_gemm_lds_kernel<<<g, 128, 0, stream>>>(A, WT, bias, addsrc, oF, oB, Mn, Nn, Kn);
    };

    // ---- x_norm (bf16) ----
    rmsnorm_bf16_kernel<<<MM, 128, 0, stream>>>(x, n1_w, xn_bf);

    // ---- neighbor affinity path ----
    gemm(xn_bf, aqT, aq_b, nullptr, aQ, nullptr, MM, DM, DM);
    gemm(xn_bf, akT, ak_b, nullptr, aK, nullptr, MM, DM, DM);
    neighbor_kernel<<<BB * (SS - 1), 32, 0, stream>>>(aQ, aK, loga);
    scan_kernel<<<BB, SS, 0, stream>>>(loga, Lb);

    // ---- q/k/v projections (bf16 outputs for attention WMMA) ----
    gemm(xn_bf, wqT, wq_b, nullptr, nullptr, qp_bf, MM, D2, DM);
    gemm(xn_bf, wkT, wk_b, nullptr, nullptr, kp_bf, MM, D2, DM);
    gemm(xn_bf, wvT, wv_b, nullptr, nullptr, v_bf, MM, DM, DM);

    // ---- differential attention (two-pass flash style, all-WMMA) ----
    dim3 ag(SS / 16, NH, BB);
    attn_stats_kernel<<<ag, 32, 0, stream>>>(qp_bf, kp_bf, pad, m1s, z1s, m2s, z2s);
    attn_ctx_kernel<<<ag, 32, 0, stream>>>(qp_bf, kp_bf, v_bf, pad,
                                           m1s, z1s, m2s, z2s, Lb, lamp, ctx);

    // ---- group norm -> bf16, out-projection with residual: h = x + ctx@out_w + b ----
    groupnorm_kernel<<<BB * NH, 256, 0, stream>>>(ctx, gng, gnb, gn_bf);
    gemm(gn_bf, outT, out_b, x, hbuf, nullptr, MM, DM, DM);

    // ---- FFN: hn = rmsnorm(h); act = silu(hn@wg)*(hn@w1); out = h + act@w2 ----
    rmsnorm_bf16_kernel<<<MM, 128, 0, stream>>>(hbuf, n2_w, hn_bf);
    {
        dim3 g(FF / 64, MM / 16);
        swiglu_gemm_kernel<<<g, 32, 0, stream>>>(hn_bf, wgT, w1T, act_bf, MM, FF, DM);
    }
    gemm(act_bf, w2T, nullptr, hbuf, out, nullptr, MM, DM, FF);
}
